// SelfAttention_56358560858582
// MI455X (gfx1250) — compile-verified, matrix operand order checked by reading, not run on hardware
//
#include <hip/hip_runtime.h>

typedef __bf16 bf16;
typedef __attribute__((ext_vector_type(8)))  __bf16 v8bf;
typedef __attribute__((ext_vector_type(16))) __bf16 v16bf;
typedef __attribute__((ext_vector_type(8)))  float  v8f;

#define R_DIM   128
#define N_DIM   256
#define D_DIM   128
#define H_DIM   8
#define DH_DIM  64
#define INNER_DIM 512
#define DP_DIM  128

__device__ __forceinline__ bf16 f2bf(float f) {
  union { float f; unsigned u; } a; a.f = f;
  unsigned r = a.u + 0x7FFFu + ((a.u >> 16) & 1u);   // round-to-nearest-even
  unsigned short h = (unsigned short)(r >> 16);
  return __builtin_bit_cast(bf16, h);
}

// ---- WMMA A-fragment load from global (ISA 7.12.2 16-bit 16x32 layout) ----
// lane 0-15: row = base+lane,    K-chunks [k0..k0+7]   and [k0+16..k0+23]
// lane16-31: row = base+lane-16, K-chunks [k0+8..k0+15] and [k0+24..k0+31]
__device__ __forceinline__ v16bf load_frag_g(const bf16* __restrict__ base,
                                             long row, long kBase, long ld, int half8) {
  const bf16* p = base + row * ld + kBase + half8;
  v8bf lo = *(const v8bf*)(p);
  v8bf hi = *(const v8bf*)(p + 16);
  v16bf r;
#pragma unroll
  for (int i = 0; i < 8; ++i) { r[i] = lo[i]; r[8 + i] = hi[i]; }
  return r;
}

// ---- Generic batched GEMM: C[z] = A[z](MxK) * Bt[z](NxK)^T with fused epilogues ----
// block = 256 threads (8 waves). Block tile 128x64: wave w -> rows [w*16, w*16+16),
// 4 sub-tiles of 16 columns. B slab (64 x 128) staged in LDS via async global->LDS
// copies (ASYNCcnt); K chunked by 128, inner steps of 32 with v_wmma_f32_16x16x32_bf16.
// Per K-step: all 4 B fragments are fetched from LDS first, then 4 WMMAs issue
// back-to-back (single dscnt wait instead of one per WMMA).
// Epilogue modes:
//  0: C[m*N+n] = acc + bias[n]                       (f32 out)
//  1: q->qw  : r=m>>8,i=m&255,h=n>>6,d=n&63; Cb[(h*256+i)*8192+r*64+d]=bf16(acc*w*0.125)
//  2: kv->k,v: n<512 -> Cb (k layout);  n>=512 -> C2 (v^T layout [r][h][d][j])
//  3: attnV->o1: r=z>>3,h=z&7; Cb[(r*256+m)*512+h*64+n]=bf16(acc)
__global__ __launch_bounds__(256)
void gemm_bf16_nt(const bf16* __restrict__ A, const bf16* __restrict__ Bt,
                  float* __restrict__ C, bf16* __restrict__ Cb, bf16* __restrict__ C2,
                  const float* __restrict__ wbuf, const float* __restrict__ bias,
                  int M, int N, int K,
                  long aBS, long bBS, long cBS, int aMod, int bMod, int mode) {
  __shared__ __attribute__((aligned(16))) bf16 Bs[64 * 128];
  int z = blockIdx.z;
  long aOff = (long)(aMod ? (z % aMod) : z) * aBS;
  long bOff = (long)(bMod ? (z % bMod) : z) * bBS;
  long cOff = (long)z * cBS;
  int lane = threadIdx.x & 31;
  int wave = threadIdx.x >> 5;
  long mBase = (long)blockIdx.x * 128 + wave * 16;
  long nBase = (long)blockIdx.y * 64;
  long aRow = mBase + (lane & 15);
  int half8 = ((lane >> 4) & 1) * 8;
  int tid = threadIdx.x;
  v8f acc[4] = {};
  for (long k0 = 0; k0 < K; k0 += 128) {
    // --- cooperative async copy of Bt[nBase..+63][k0..+127] into LDS (16KB) ---
#pragma unroll
    for (int u = 0; u < 4; ++u) {
      int idx = tid + 256 * u;           // 1024 x 16B chunks
      int row = idx >> 4;                // 16 chunks per 128-elem row
      int c16 = idx & 15;
      const bf16* src = Bt + bOff + (nBase + row) * (long)K + k0 + c16 * 8;
      unsigned dst = (unsigned)(unsigned long long)(const void*)(Bs + row * 128 + c16 * 8);
      asm volatile("global_load_async_to_lds_b128 %0, %1, off"
                   :: "v"(dst), "v"(src) : "memory");
    }
    asm volatile("s_wait_asynccnt 0x0" ::: "memory");
    __syncthreads();
#pragma unroll
    for (int ks = 0; ks < 4; ++ks) {
      long k = k0 + ks * 32;
      v16bf a = load_frag_g(A + aOff, aRow, k, K, half8);
      v16bf b[4];
#pragma unroll
      for (int t = 0; t < 4; ++t) {      // issue all 8 ds_load_b128 first
        const bf16* bp = Bs + (t * 16 + (lane & 15)) * 128 + ks * 32 + half8;
        v8bf lo = *(const v8bf*)(bp);
        v8bf hi = *(const v8bf*)(bp + 16);
#pragma unroll
        for (int i = 0; i < 8; ++i) { b[t][i] = lo[i]; b[t][8 + i] = hi[i]; }
      }
#pragma unroll
      for (int t = 0; t < 4; ++t)        // then 4 WMMAs back-to-back
        acc[t] = __builtin_amdgcn_wmma_f32_16x16x32_bf16(
            false, a, false, b[t], (short)0, acc[t], false, false);
    }
    __syncthreads();
  }
  int nLane = lane & 15;
  int mHi = half8;                         // same 8-row shift for C layout
#pragma unroll
  for (int t = 0; t < 4; ++t) {
    long n = nBase + t * 16 + nLane;
#pragma unroll
    for (int i = 0; i < 8; ++i) {
      long m = mBase + (long)i + mHi;
      float val = acc[t][i];
      if (mode == 0) {
        float bv = bias ? bias[n] : 0.0f;
        C[cOff + m * (long)N + n] = val + bv;
      } else if (mode == 1) {
        int r = (int)(m >> 8), ii = (int)(m & 255);
        int h = (int)(n >> 6), d = (int)(n & 63);
        float wv = wbuf[((long)ii * 8 + h) * 128 + r];
        Cb[((long)h * N_DIM + ii) * 8192 + (long)r * 64 + d] = f2bf(val * wv * 0.125f);
      } else if (mode == 2) {
        int r = (int)(m >> 8), ii = (int)(m & 255);
        if (n < 512) {
          int h = (int)(n >> 6), d = (int)(n & 63);
          Cb[((long)h * N_DIM + ii) * 8192 + (long)r * 64 + d] = f2bf(val);
        } else {
          int c = (int)(n - 512), h = c >> 6, d = c & 63;
          C2[(((long)r * 8 + h) * 64 + d) * N_DIM + ii] = f2bf(val);
        }
      } else { // mode 3
        int r = z >> 3, h = z & 7;
        Cb[((long)(r * N_DIM + m)) * INNER_DIM + h * DH_DIM + n] = f2bf(val);
      }
    }
  }
}

// ---- elementwise f32 -> bf16 ----
__global__ void k_f32_to_bf16(const float* __restrict__ in, bf16* __restrict__ out, long n) {
  long i = (long)blockIdx.x * blockDim.x + threadIdx.x;
  if (i < n) out[i] = f2bf(in[i]);
}

// ---- transpose [K][N] f32 -> [N][K] bf16 (for B^T operands) ----
__global__ void k_transpose_bf16(const float* __restrict__ in, bf16* __restrict__ out,
                                 int K, int N) {
  long t = (long)blockIdx.x * blockDim.x + threadIdx.x;
  if (t >= (long)K * N) return;
  int k = (int)(t / N), n = (int)(t % N);
  out[(long)n * K + k] = f2bf(in[t]);
}

// ---- PositionalWiseWeight: per-n block, compute qsw, logits over (h,r), softmax over r ----
__global__ __launch_bounds__(256)
void k_seqweight(const float* __restrict__ x, const float* __restrict__ swq_w,
                 const float* __restrict__ swq_b, const float* __restrict__ ksw,
                 const float* __restrict__ swk_b, float* __restrict__ wbuf) {
  __shared__ float qswL[128];
  __shared__ float logitsL[1024];
  int n = blockIdx.x;
  int t = threadIdx.x;
  if (t < 128) {                       // qsw[n, c] = x[r=0, n, :] @ swq_w[:, c] + b
    float s = swq_b[t];
    const float* xr = x + (long)n * D_DIM;   // r = 0 row
    for (int d = 0; d < 128; ++d) s += xr[d] * swq_w[d * 128 + t];
    qswL[t] = s;
  }
  __syncthreads();
#pragma unroll
  for (int u = 0; u < 4; ++u) {        // logits[h][r], 1024 total, 4 per thread
    int idx = t * 4 + u;
    int h = idx >> 7, r = idx & 127;
    const float* kr = ksw + ((long)r * N_DIM + n) * D_DIM + h * 16;
    const float* qh = qswL + h * 16;
    const float* bb = swk_b + h * 16;
    float s = 0.f;
#pragma unroll
    for (int d = 0; d < 16; ++d) s += qh[d] * (kr[d] + bb[d]);
    logitsL[idx] = 0.25f * s;          // 1/sqrt(SWD=16)
  }
  __syncthreads();
  int wave = t >> 5, lane = t & 31;    // wave w -> head w; softmax over r (128)
  float v0 = logitsL[wave * 128 + lane];
  float v1 = logitsL[wave * 128 + lane + 32];
  float v2 = logitsL[wave * 128 + lane + 64];
  float v3 = logitsL[wave * 128 + lane + 96];
  float mx = fmaxf(fmaxf(v0, v1), fmaxf(v2, v3));
  for (int o = 16; o >= 1; o >>= 1) mx = fmaxf(mx, __shfl_xor(mx, o, 32));
  float e0 = __expf(v0 - mx), e1 = __expf(v1 - mx);
  float e2 = __expf(v2 - mx), e3 = __expf(v3 - mx);
  float sm = e0 + e1 + e2 + e3;
  for (int o = 16; o >= 1; o >>= 1) sm += __shfl_xor(sm, o, 32);
  float inv = 1.0f / sm;
  float* wr = wbuf + ((long)n * 8 + wave) * 128;
  wr[lane] = e0 * inv; wr[lane + 32] = e1 * inv;
  wr[lane + 64] = e2 * inv; wr[lane + 96] = e3 * inv;
}

// ---- pair bias: LayerNorm(128ch) + Linear(128->8), one wave per (i,j) ----
__global__ __launch_bounds__(256)
void k_pair_bias(const float* __restrict__ pair, const float* __restrict__ g,
                 const float* __restrict__ bta, const float* __restrict__ wp,
                 float* __restrict__ biasbuf) {
  int wave = threadIdx.x >> 5, lane = threadIdx.x & 31;
  long pos = (long)blockIdx.x * 8 + wave;
  int i = (int)(pos >> 8), j = (int)(pos & 255);
  const float4 v4 = *(const float4*)(pair + pos * DP_DIM + lane * 4);
  float s  = v4.x + v4.y + v4.z + v4.w;
  float sq = v4.x * v4.x + v4.y * v4.y + v4.z * v4.z + v4.w * v4.w;
  for (int o = 16; o >= 1; o >>= 1) { s += __shfl_xor(s, o, 32); sq += __shfl_xor(sq, o, 32); }
  float mean = s * (1.0f / 128.0f);
  float var  = sq * (1.0f / 128.0f) - mean * mean;
  float rstd = rsqrtf(var + 1e-5f);
  int c0 = lane * 4;
  float ln0 = (v4.x - mean) * rstd * g[c0 + 0] + bta[c0 + 0];
  float ln1 = (v4.y - mean) * rstd * g[c0 + 1] + bta[c0 + 1];
  float ln2 = (v4.z - mean) * rstd * g[c0 + 2] + bta[c0 + 2];
  float ln3 = (v4.w - mean) * rstd * g[c0 + 3] + bta[c0 + 3];
  float hb[8];
#pragma unroll
  for (int h = 0; h < 8; ++h)
    hb[h] = ln0 * wp[(c0 + 0) * 8 + h] + ln1 * wp[(c0 + 1) * 8 + h] +
            ln2 * wp[(c0 + 2) * 8 + h] + ln3 * wp[(c0 + 3) * 8 + h];
#pragma unroll
  for (int h = 0; h < 8; ++h)
    for (int o = 16; o >= 1; o >>= 1) hb[h] += __shfl_xor(hb[h], o, 32);
  if (lane == 0) {
#pragma unroll
    for (int h = 0; h < 8; ++h)
      biasbuf[((long)h * N_DIM + i) * N_DIM + j] = hb[h];
  }
}

// ---- attention softmax over j; write bf16 attn ----
__global__ __launch_bounds__(256)
void k_softmax(const float* __restrict__ dots, const float* __restrict__ biasbuf,
               bf16* __restrict__ attn) {
  __shared__ float red[256];
  int row = blockIdx.x;        // h*256 + i
  int j = threadIdx.x;
  long idx = (long)row * N_DIM + j;
  float v = dots[idx] + biasbuf[idx];
  red[j] = v; __syncthreads();
  for (int s = 128; s > 0; s >>= 1) { if (j < s) red[j] = fmaxf(red[j], red[j + s]); __syncthreads(); }
  float m = red[0]; __syncthreads();
  float e = __expf(v - m);
  red[j] = e; __syncthreads();
  for (int s = 128; s > 0; s >>= 1) { if (j < s) red[j] += red[j + s]; __syncthreads(); }
  float inv = 1.0f / red[0];
  attn[idx] = f2bf(e * inv);
}

extern "C" void kernel_launch(void* const* d_in, const int* in_sizes, int n_in,
                              void* d_out, int out_size, void* d_ws, size_t ws_size,
                              hipStream_t stream) {
  (void)in_sizes; (void)n_in; (void)out_size; (void)ws_size;
  const float* x     = (const float*)d_in[0];
  const float* pair  = (const float*)d_in[1];
  const float* wq    = (const float*)d_in[2];
  const float* wkv   = (const float*)d_in[3];
  const float* wo    = (const float*)d_in[4];
  const float* bo    = (const float*)d_in[5];
  const float* pn_g  = (const float*)d_in[6];
  const float* pn_b  = (const float*)d_in[7];
  const float* wp    = (const float*)d_in[8];
  const float* swq_w = (const float*)d_in[9];
  const float* swq_b = (const float*)d_in[10];
  const float* swk_w = (const float*)d_in[11];
  const float* swk_b = (const float*)d_in[12];
  float* out = (float*)d_out;

  char* ws = (char*)d_ws;
  size_t off = 0;
  auto alloc = [&](size_t bytes) -> char* {
    char* p = ws + off; off += (bytes + 255) & ~(size_t)255; return p;
  };
  const long MROW = (long)R_DIM * N_DIM;   // 32768
  bf16* x_bf   = (bf16*)alloc(MROW * 128 * 2);
  bf16* wq_t   = (bf16*)alloc(512 * 128 * 2);
  bf16* wkv_t  = (bf16*)alloc(1024 * 128 * 2);
  bf16* swk_t  = (bf16*)alloc(128 * 128 * 2);
  bf16* wo_t   = (bf16*)alloc(128 * 512 * 2);
  float* ksw_f = (float*)alloc(MROW * 128 * 4);
  float* wbuf  = (float*)alloc((size_t)N_DIM * 8 * 128 * 4);
  bf16* qw_bf  = (bf16*)alloc((size_t)8 * 256 * 8192 * 2);
  bf16* k_bf   = (bf16*)alloc((size_t)8 * 256 * 8192 * 2);
  bf16* v_t    = (bf16*)alloc((size_t)1024 * 64 * 256 * 2);
  float* dots  = (float*)alloc((size_t)8 * 256 * 256 * 4);
  float* biasb = (float*)alloc((size_t)8 * 256 * 256 * 4);
  bf16* attn_bf= (bf16*)alloc((size_t)8 * 256 * 256 * 2);
  bf16* o1_bf  = (bf16*)alloc(MROW * 512 * 2);

  dim3 blk(256);
  // 1) converts / weight transposes
  k_f32_to_bf16<<<(unsigned)((MROW * 128 + 255) / 256), blk, 0, stream>>>(x, x_bf, MROW * 128);
  k_transpose_bf16<<<(128 * 512 + 255) / 256, blk, 0, stream>>>(wq, wq_t, 128, 512);
  k_transpose_bf16<<<(128 * 1024 + 255) / 256, blk, 0, stream>>>(wkv, wkv_t, 128, 1024);
  k_transpose_bf16<<<(128 * 128 + 255) / 256, blk, 0, stream>>>(swk_w, swk_t, 128, 128);
  k_transpose_bf16<<<(512 * 128 + 255) / 256, blk, 0, stream>>>(wo, wo_t, 512, 128);
  // 2) ksw projection (f32, consumed by seq-weight kernel)
  gemm_bf16_nt<<<dim3(256, 2, 1), blk, 0, stream>>>(x_bf, swk_t, ksw_f, nullptr, nullptr,
      nullptr, nullptr, 32768, 128, 128, 0, 0, 0, 0, 0, 0);
  // 3) soft row weights (softmax over r)
  k_seqweight<<<256, blk, 0, stream>>>(x, swq_w, swq_b, ksw_f, swk_b, wbuf);
  // 4) q projection with fused w*DH^-0.5 scaling -> qw_bf in WMMA layout
  gemm_bf16_nt<<<dim3(256, 8, 1), blk, 0, stream>>>(x_bf, wq_t, nullptr, qw_bf, nullptr,
      wbuf, nullptr, 32768, 512, 128, 0, 0, 0, 0, 0, 1);
  // 5) kv projection with fused scatter -> k_bf ([h][j][r*64+d]) and v_t ([r][h][d][j])
  gemm_bf16_nt<<<dim3(256, 16, 1), blk, 0, stream>>>(x_bf, wkv_t, nullptr, k_bf, v_t,
      nullptr, nullptr, 32768, 1024, 128, 0, 0, 0, 0, 0, 2);
  // 6) tied dots: per-head 256x256, K = R*DH = 8192
  gemm_bf16_nt<<<dim3(2, 4, 8), blk, 0, stream>>>(qw_bf, k_bf, dots, nullptr, nullptr,
      nullptr, nullptr, 256, 256, 8192, 256L * 8192, 256L * 8192, 256L * 256, 0, 0, 0);
  // 7) pair bias (LN + proj)
  k_pair_bias<<<8192, blk, 0, stream>>>(pair, pn_g, pn_b, wp, biasb);
  // 8) softmax -> bf16 attn
  k_softmax<<<2048, blk, 0, stream>>>(dots, biasb, attn_bf);
  // 9) attn @ v: batched over (r,h)=1024, attn shared across r (aMod=8),
  //    epilogue writes o1_bf directly in [(r,i)][h*64+d] layout
  gemm_bf16_nt<<<dim3(2, 1, 1024), blk, 0, stream>>>(attn_bf, v_t, nullptr, o1_bf, nullptr,
      nullptr, nullptr, 256, 64, 256, 256L * 256, 64L * 256, 0, 8, 0, 3);
  // 10) final projection with bias -> d_out (f32)
  gemm_bf16_nt<<<dim3(256, 2, 1), blk, 0, stream>>>(o1_bf, wo_t, out, nullptr, nullptr,
      nullptr, bo, 32768, 128, 512, 0, 0, 0, 0, 0, 0);
}
